// PointNetSetAbstractionVotes_40140764348416
// MI455X (gfx1250) — matrix operand order repaired and checked by reading, not compiled, not run on hardware
//
#include <hip/hip_runtime.h>

typedef __attribute__((ext_vector_type(16))) __bf16 v16bf;
typedef __attribute__((ext_vector_type(8)))  float  v8f;

#define B_   8
#define N_   16384
#define S_   1024
#define NS_  64
#define D_   128
#define M_   (B_*S_*NS_)        /* 524288 */
#define R2_  0.04f
#define EPS_ 1e-5f

/* ---- workspace layout (bytes) ---- */
#define IDX_OFF    0ull                       /* B*S*NS int32            = 2 MB    */
#define FEATS_OFF  2097152ull                 /* B*N*128 bf16            = 32 MB   */
#define W_OFF      35651584ull                /* swizzled weights        = 136 KB  */
#define W0_OFF     (W_OFF)                    /* 5*8*1024  = 40960 B  */
#define W1_OFF     (W_OFF + 40960ull)         /* 4*8*1024  = 32768 B  */
#define W2_OFF     (W_OFF + 73728ull)         /* 4*16*1024 = 65536 B  */
#define STAT_OFF   35790848ull                /* l0:256f, l1:256f, l2:512f = 4 KB  */
#define SCSH_OFF   35794944ull                /* sc/sh per layer, 1024 f32 = 4 KB  */
#define Y1_OFF     35799040ull                /* M*128 bf16 = 128 MB  */
#define Y2_OFF     (Y1_OFF + 134217728ull)    /* M*256 bf16 = 256 MB  */
#define Y0_OFF     Y2_OFF                     /* y0 overlaps y2 (dead before GEMM2 writes) */

/* ---- output layout (float elements) ---- */
#define OUT_XYZ  0
#define OUT_FEAT 24576
#define OUT_FPS  (24576 + 2097152)

__device__ __forceinline__ unsigned short f2bf(float f) {
    unsigned u = __float_as_uint(f);
    unsigned r = (u + 0x7fffu + ((u >> 16) & 1u)) >> 16;   /* RNE */
    return (unsigned short)r;
}
__device__ __forceinline__ float bf2f(unsigned short h) {
    return __uint_as_float(((unsigned)h) << 16);
}

union Frag {
    uint4          q[2];
    unsigned short h[16];
    v16bf          v;
};

/* =====================================================================
 * 1. Ball query: one wave per query, ordered early-exit scan.
 * ===================================================================== */
__global__ __launch_bounds__(256) void k_ballquery(const float* __restrict__ xyz,
                                                   const int*   __restrict__ fps,
                                                   int*         __restrict__ idxb,
                                                   float*       __restrict__ out)
{
    int wave = threadIdx.x >> 5, lane = threadIdx.x & 31;
    int q = blockIdx.x * 8 + wave;
    if (q >= B_ * S_) return;
    int b = q >> 10, s = q & (S_ - 1);
    int fi = fps[b * S_ + s];
    const float* xb = xyz + (size_t)b * N_ * 3;
    float qx = xb[fi * 3 + 0], qy = xb[fi * 3 + 1], qz = xb[fi * 3 + 2];
    if (lane == 0) {
        out[OUT_XYZ + q * 3 + 0] = qx;
        out[OUT_XYZ + q * 3 + 1] = qy;
        out[OUT_XYZ + q * 3 + 2] = qz;
        out[OUT_FPS + q] = __int_as_float(fi);
    }
    int count = 0, first = fi;
    int* myidx = idxb + (size_t)q * NS_;
    for (int base = 0; base < N_; base += 32) {
        int n = base + lane;
        float dx = xb[n * 3 + 0] - qx;
        float dy = xb[n * 3 + 1] - qy;
        float dz = xb[n * 3 + 2] - qz;
        bool inb = (dx * dx + dy * dy + dz * dz) <= R2_;
        unsigned mask = __builtin_amdgcn_ballot_w32(inb);
        int pos = count + __popc(mask & ((1u << lane) - 1u));
        if (inb && pos < NS_) myidx[pos] = n;
        if (count == 0 && mask != 0u) first = base + (__ffs(mask) - 1);
        count += __popc(mask);
        if (count >= NS_) break;
    }
    for (int p = count + lane; p < NS_; p += 32) myidx[p] = first;
}

/* =====================================================================
 * 2. points [B,D,N] f32  ->  featsT [B,N,D] bf16  (LDS tile transpose)
 * ===================================================================== */
__global__ __launch_bounds__(256) void k_transpose(const float* __restrict__ pts,
                                                   unsigned short* __restrict__ ft)
{
    __shared__ float t[32][33];
    int bi = blockIdx.x;
    int nt = bi & (N_ / 32 - 1);
    int dt = (bi >> 9) & 3;
    int b  = bi >> 11;
    int r = threadIdx.x >> 5, c = threadIdx.x & 31;
    const float* src = pts + ((size_t)b * D_ + dt * 32) * N_ + nt * 32;
    for (int rr = r; rr < 32; rr += 8) t[rr][c] = src[(size_t)rr * N_ + c];
    __syncthreads();
    unsigned short* dst = ft + ((size_t)b * N_ + nt * 32) * D_ + dt * 32;
    for (int rr = r; rr < 32; rr += 8) dst[(size_t)rr * D_ + c] = f2bf(t[c][rr]);
}

/* =====================================================================
 * 3. Prep: zero stats + pre-swizzle weights into B-fragment order.
 *    Fragment (kt,nt): halves[lane*16+e] = W[k = kt*32 + (lane>>4)*8 + (e&7) + (e>>3)*16]
 *                                          [n = nt*16 + (lane&15)]
 *    Layer0 column permutation: k<128 -> in 3+k ; k=128..130 -> xyz ; pad 0.
 * ===================================================================== */
__global__ __launch_bounds__(256) void k_prep(const float* __restrict__ W0,
                                              const float* __restrict__ W1,
                                              const float* __restrict__ W2,
                                              unsigned short* __restrict__ w0s,
                                              unsigned short* __restrict__ w1s,
                                              unsigned short* __restrict__ w2s,
                                              float* __restrict__ stats)
{
    int gt = blockIdx.x * 256 + threadIdx.x;
    int gs = gridDim.x * 256;
    for (int i = gt; i < 1024; i += gs) stats[i] = 0.f;
    /* layer 0: KT=5, NT=8, cin=131 (permuted+padded to 160), cout=128 */
    for (int i = gt; i < 5 * 8 * 512; i += gs) {
        int frag = i >> 9, w = i & 511, lane = w >> 4, e = w & 15;
        int kt = frag >> 3;
        int n  = (frag & 7) * 16 + (lane & 15);
        int k  = kt * 32 + ((lane >> 4) << 3) + (e & 7) + ((e >> 3) << 4);
        float v = 0.f;
        if (k < 128)      v = W0[n * 131 + 3 + k];
        else if (k < 131) v = W0[n * 131 + (k - 128)];
        w0s[i] = f2bf(v);
    }
    /* layer 1: KT=4, NT=8, 128x128 */
    for (int i = gt; i < 4 * 8 * 512; i += gs) {
        int frag = i >> 9, w = i & 511, lane = w >> 4, e = w & 15;
        int kt = frag >> 3;
        int n  = (frag & 7) * 16 + (lane & 15);
        int k  = kt * 32 + ((lane >> 4) << 3) + (e & 7) + ((e >> 3) << 4);
        w1s[i] = f2bf(W1[n * 128 + k]);
    }
    /* layer 2: KT=4, NT=16, 128 -> 256 */
    for (int i = gt; i < 4 * 16 * 512; i += gs) {
        int frag = i >> 9, w = i & 511, lane = w >> 4, e = w & 15;
        int kt = frag >> 4;
        int n  = (frag & 15) * 16 + (lane & 15);
        int k  = kt * 32 + ((lane >> 4) << 3) + (e & 7) + ((e >> 3) << 4);
        w2s[i] = f2bf(W2[n * 128 + k]);
    }
}

/* =====================================================================
 * 4. GEMM layer: bf16 WMMA, fused gather (L0) / BN+ReLU (L>0),
 *    bf16 output + per-channel sum/sumsq accumulation.
 * ===================================================================== */
template <int LAYER>
__global__ __launch_bounds__(256) void k_gemm(const unsigned short* __restrict__ xin,
                                              const int*   __restrict__ idxb,
                                              const float* __restrict__ xyz,
                                              const float* __restrict__ newxyz,
                                              const uint4* __restrict__ wswz,
                                              const float* __restrict__ sc,
                                              const float* __restrict__ sh,
                                              unsigned short* __restrict__ yout,
                                              float* __restrict__ gstat)
{
    constexpr int KT  = (LAYER == 0) ? 5 : 4;
    constexpr int NT  = (LAYER == 2) ? 16 : 8;
    constexpr int NCH = NT * 16;
    constexpr int NW4 = KT * NT * 64;            /* uint4 count of swizzled weights */

    __shared__ uint4 sW[NW4];
    __shared__ float sSum[NCH], sSsq[NCH];
    __shared__ float sSc[128], sSh[128];

    int tid = threadIdx.x;
    for (int i = tid; i < NW4; i += 256) sW[i] = wswz[i];
    for (int i = tid; i < NCH; i += 256) { sSum[i] = 0.f; sSsq[i] = 0.f; }
    if constexpr (LAYER != 0) {
        for (int i = tid; i < 128; i += 256) { sSc[i] = sc[i]; sSh[i] = sh[i]; }
    }
    __syncthreads();

    int wave = tid >> 5, lane = tid & 31;
    int lrow = lane & 15;
    int K0h  = (lane >> 4) << 3;                 /* 0 or 8 */

    for (int t = blockIdx.x * 8 + wave; t < M_ / 16; t += gridDim.x * 8) {
        int myrow = t * 16 + lrow;
        size_t aBase = 0;
        float rx = 0.f, ry = 0.f, rz = 0.f;
        if constexpr (LAYER == 0) {
            int n = idxb[myrow];
            int b = myrow >> 16;                 /* S*NS = 65536 rows per batch */
            int q = myrow >> 6;
            size_t pn = (size_t)(b * N_ + n);
            aBase = pn * 128;
            rx = xyz[pn * 3 + 0] - newxyz[q * 3 + 0];
            ry = xyz[pn * 3 + 1] - newxyz[q * 3 + 1];
            rz = xyz[pn * 3 + 2] - newxyz[q * 3 + 2];
        } else {
            aBase = (size_t)myrow * 128;
        }

        v8f acc[NT];
        #pragma unroll
        for (int nt = 0; nt < NT; ++nt)
            #pragma unroll
            for (int i = 0; i < 8; ++i) acc[nt][i] = 0.f;

        #pragma unroll
        for (int kt = 0; kt < KT; ++kt) {
            Frag a;
            int K0 = kt * 32 + K0h;
            if (LAYER == 0 && kt == 4) {         /* synthesized xyz-rel + zero pad tile */
                a.q[0] = make_uint4(0u, 0u, 0u, 0u);
                a.q[1] = make_uint4(0u, 0u, 0u, 0u);
                if (lane < 16) { a.h[0] = f2bf(rx); a.h[1] = f2bf(ry); a.h[2] = f2bf(rz); }
            } else {
                const uint4* ap = (const uint4*)(xin + aBase + K0);
                a.q[0] = ap[0];                  /* halves K0   .. K0+7  */
                a.q[1] = ap[2];                  /* halves K0+16.. K0+23 */
                if constexpr (LAYER != 0) {      /* fused BN + ReLU on input */
                    #pragma unroll
                    for (int e = 0; e < 16; ++e) {
                        int k = K0 + e + ((e >= 8) ? 8 : 0);
                        float f = bf2f(a.h[e]);
                        f = fmaf(sSc[k], f, sSh[k]);
                        f = fmaxf(f, 0.f);
                        a.h[e] = f2bf(f);
                    }
                }
            }
            #pragma unroll
            for (int nt = 0; nt < NT; ++nt) {
                Frag bw;
                const uint4* bp = sW + (kt * NT + nt) * 64 + lane * 2;
                bw.q[0] = bp[0];
                bw.q[1] = bp[1];
                acc[nt] = __builtin_amdgcn_wmma_f32_16x16x32_bf16(
                    false, a.v, false, bw.v, (short)0, acc[nt], false, false);
            }
        }

        /* epilogue: C layout — VGPR r: lanes0-15 M=r, lanes16-31 M=r+8; N = lane&15 */
        int row0 = t * 16 + ((lane >> 4) << 3);
        #pragma unroll
        for (int nt = 0; nt < NT; ++nt) {
            int c = nt * 16 + (lane & 15);
            float s1 = 0.f, s2 = 0.f;
            #pragma unroll
            for (int r = 0; r < 8; ++r) {
                float v = acc[nt][r];
                s1 += v; s2 += v * v;
                yout[(size_t)(row0 + r) * NCH + c] = f2bf(v);
            }
            unsafeAtomicAdd(&sSum[c], s1);
            unsafeAtomicAdd(&sSsq[c], s2);
        }
    }

    __syncthreads();
    for (int c = tid; c < NCH; c += 256) {
        unsafeAtomicAdd(&gstat[c], sSum[c]);
        unsafeAtomicAdd(&gstat[NCH + c], sSsq[c]);
    }
}

/* =====================================================================
 * 5. Fold BN stats into per-channel scale/shift.
 * ===================================================================== */
__global__ __launch_bounds__(256) void k_stats(const float* __restrict__ stat,
                                               const float* __restrict__ g,
                                               const float* __restrict__ be,
                                               int nch,
                                               float* __restrict__ scale,
                                               float* __restrict__ shift)
{
    int c = blockIdx.x * 256 + threadIdx.x;
    if (c >= nch) return;
    float inv = 1.0f / (float)M_;
    float mean = stat[c] * inv;
    float var  = stat[nch + c] * inv - mean * mean;
    float rs   = rsqrtf(var + EPS_);
    float scl  = g[c] * rs;
    scale[c] = scl;
    shift[c] = be[c] - mean * scl;
}

/* =====================================================================
 * 6. BN + ReLU + max over NS, write [B,256,S].
 * ===================================================================== */
__global__ __launch_bounds__(256) void k_maxpool(const unsigned short* __restrict__ y2,
                                                 const float* __restrict__ scale,
                                                 const float* __restrict__ shift,
                                                 float* __restrict__ out)
{
    int q = blockIdx.x, c = threadIdx.x;
    float scl = scale[c], shf = shift[c];
    const unsigned short* yp = y2 + (size_t)q * NS_ * 256 + c;
    float m = 0.f;                               /* ReLU output >= 0 */
    for (int j = 0; j < NS_; ++j) {
        float v = fmaf(scl, bf2f(yp[(size_t)j * 256]), shf);
        v = fmaxf(v, 0.f);
        m = fmaxf(m, v);
    }
    int b = q >> 10, s = q & (S_ - 1);
    out[OUT_FEAT + ((size_t)b * 256 + c) * S_ + s] = m;
}

extern "C" void kernel_launch(void* const* d_in, const int* in_sizes, int n_in,
                              void* d_out, int out_size, void* d_ws, size_t ws_size,
                              hipStream_t stream)
{
    const float* xyz    = (const float*)d_in[0];
    const float* points = (const float*)d_in[1];
    const int*   fps    = (const int*)d_in[2];
    const float* W0  = (const float*)d_in[3];
    const float* g0  = (const float*)d_in[5];
    const float* be0 = (const float*)d_in[6];
    const float* W1  = (const float*)d_in[7];
    const float* g1  = (const float*)d_in[9];
    const float* be1 = (const float*)d_in[10];
    const float* W2  = (const float*)d_in[11];
    const float* g2  = (const float*)d_in[13];
    const float* be2 = (const float*)d_in[14];
    float* out = (float*)d_out;

    char* ws = (char*)d_ws;
    int*            idxb  = (int*)(ws + IDX_OFF);
    unsigned short* feats = (unsigned short*)(ws + FEATS_OFF);
    unsigned short* w0s   = (unsigned short*)(ws + W0_OFF);
    unsigned short* w1s   = (unsigned short*)(ws + W1_OFF);
    unsigned short* w2s   = (unsigned short*)(ws + W2_OFF);
    float*          stats = (float*)(ws + STAT_OFF);
    float*          scsh  = (float*)(ws + SCSH_OFF);
    unsigned short* y0    = (unsigned short*)(ws + Y0_OFF);
    unsigned short* y1    = (unsigned short*)(ws + Y1_OFF);
    unsigned short* y2    = (unsigned short*)(ws + Y2_OFF);

    k_prep<<<64, 256, 0, stream>>>(W0, W1, W2, w0s, w1s, w2s, stats);
    k_ballquery<<<1024, 256, 0, stream>>>(xyz, fps, idxb, out);
    k_transpose<<<16384, 256, 0, stream>>>(points, feats);

    k_gemm<0><<<2048, 256, 0, stream>>>(feats, idxb, xyz, out + OUT_XYZ,
                                        (const uint4*)w0s, nullptr, nullptr,
                                        y0, stats);
    k_stats<<<1, 256, 0, stream>>>(stats, g0, be0, 128, scsh, scsh + 128);

    k_gemm<1><<<2048, 256, 0, stream>>>(y0, nullptr, nullptr, nullptr,
                                        (const uint4*)w1s, scsh, scsh + 128,
                                        y1, stats + 256);
    k_stats<<<1, 256, 0, stream>>>(stats + 256, g1, be1, 128, scsh + 256, scsh + 384);

    k_gemm<2><<<2048, 256, 0, stream>>>(y1, nullptr, nullptr, nullptr,
                                        (const uint4*)w2s, scsh + 256, scsh + 384,
                                        y2, stats + 512);
    k_stats<<<1, 256, 0, stream>>>(stats + 512, g2, be2, 256, scsh + 512, scsh + 768);

    k_maxpool<<<8192, 256, 0, stream>>>(y2, scsh + 512, scsh + 768, out);
}
